// ElementWiseGRU_15934328668948
// MI455X (gfx1250) — matrix operand order repaired and checked by reading, not compiled, not run on hardware
//
#include <hip/hip_runtime.h>

typedef __attribute__((ext_vector_type(16))) _Float16 v16h;
typedef __attribute__((ext_vector_type(8)))  float    v8f;

#define HDIM   32
#define NLAYER 2
#define NROWS  (256*1024)       /* A*B = 262144 rows */
#define LAM1   0.01f
#define WPB    8                /* waves per block (wave32) */
#define RPW    16               /* rows per wave            */
#define RPB    (WPB*RPW)
#define NTILES 26               /* 12 gru_wi + 12 gru_wh + 2 w_out1 (16-col B tiles) */

__device__ __forceinline__ v8f wmma_cf(v16h a, v16h b, v8f c) {
  // D = A(16x32 f16) x B(32x16 f16) + C(16x16 f32)
  return __builtin_amdgcn_wmma_f32_16x16x32_f16(false, a, false, b, (short)0, c, false, false);
}

__device__ __forceinline__ void wait_lds() {
  __asm__ volatile("s_wait_dscnt 0" ::: "memory");
}

// Build WMMA A-matrix (16x32 f16) regs from a per-wave LDS buffer laid out [row][col].
// A layout (16-bit): lane L holds row M=L%16; element e maps to K = (e/8)*16 + (L/16)*8 + e%8.
__device__ __forceinline__ void buildA(const float2 (*buf)[33], int lane, v16h* ar, v16h* ai) {
  const int m  = lane & 15;
  const int lh = lane >> 4;
  v16h r, i;
#pragma unroll
  for (int e = 0; e < 16; ++e) {
    const int k = ((e >> 3) << 4) + (lh << 3) + (e & 7);
    const float2 v = buf[m][k];
    r[e] = (_Float16)v.x;
    i[e] = (_Float16)v.y;
  }
  *ar = r; *ai = i;
}

__global__ __launch_bounds__(WPB*32) void gru_pipeline_kernel(
    const float* __restrict__ px,   const float* __restrict__ pe1,  const float* __restrict__ pe2,
    const float* __restrict__ ph,   const float* __restrict__ pwin, const float* __restrict__ pbin,
    const float* __restrict__ pw1,  const float* __restrict__ pbo1, const float* __restrict__ pw2,
    const float* __restrict__ pbo2, const float* __restrict__ pgwi, const float* __restrict__ pgwh,
    const float* __restrict__ pgb,  float* __restrict__ pout)
{
  // Weight B-tiles in WMMA B-layout, parts: 0=Re(W), 1=Im(W), 2=-Im(W).
  // tiles 0..11: gru_wi[l*6+ct], 12..23: gru_wh[l*6+ct], 24..25: w_out1[ct]
  __shared__ v16h   wt[NTILES][3][32];
  __shared__ float2 s_win[3][HDIM];
  __shared__ float2 s_bin[HDIM];
  __shared__ float2 s_gb[NLAYER][3*HDIM];
  __shared__ float2 s_bo1[HDIM];
  __shared__ float2 s_w2[HDIM];
  __shared__ float2 s_bo2;
  __shared__ float2 hbuf[WPB][16][33];   // per-wave: h[l] tile, C-layout (row,col)
  __shared__ float2 rbuf[WPB][16][33];   // per-wave: inp / r.h / h_new staging

  const int tid = threadIdx.x;

  // ---- cooperative weight prep: complex64 -> f16 B-tiles (B layout: K=(e/8)*16+(lane/16)*8+e%8, N=lane%16)
  for (int idx = tid; idx < NTILES*512; idx += WPB*32) {
    const int t  = idx >> 9;
    const int ln = (idx >> 4) & 31;
    const int e  = idx & 15;
    const int k  = ((e >> 3) << 4) + ((ln >> 4) << 3) + (e & 7);
    const int n  = ln & 15;
    const float* src;
    if (t < 12) {
      const int l = t / 6, ct = t % 6;
      src = pgwi + 2*(((size_t)(l*HDIM + k))*(3*HDIM) + ct*16 + n);
    } else if (t < 24) {
      const int t2 = t - 12, l = t2 / 6, ct = t2 % 6;
      src = pgwh + 2*(((size_t)(l*HDIM + k))*(3*HDIM) + ct*16 + n);
    } else {
      const int ct = t - 24;
      src = pw1 + 2*((size_t)k*HDIM + ct*16 + n);
    }
    const float re = src[0], im = src[1];
    wt[t][0][ln][e] = (_Float16)re;
    wt[t][1][ln][e] = (_Float16)im;
    wt[t][2][ln][e] = (_Float16)(-im);
  }
  for (int i = tid; i < 3*HDIM; i += WPB*32)
    s_win[i/HDIM][i%HDIM] = make_float2(pwin[2*i], pwin[2*i+1]);
  for (int i = tid; i < HDIM; i += WPB*32) {
    s_bin[i] = make_float2(pbin[2*i], pbin[2*i+1]);
    s_bo1[i] = make_float2(pbo1[2*i], pbo1[2*i+1]);
    s_w2[i]  = make_float2(pw2[2*i],  pw2[2*i+1]);
  }
  for (int i = tid; i < NLAYER*3*HDIM; i += WPB*32)
    s_gb[i/(3*HDIM)][i%(3*HDIM)] = make_float2(pgb[2*i], pgb[2*i+1]);
  if (tid == 0) s_bo2 = make_float2(pbo2[0], pbo2[1]);
  __syncthreads();

  const int lane = tid & 31;
  const int wv   = tid >> 5;
  const int row0 = (blockIdx.x * WPB + wv) * RPW;
  const int m    = lane & 15;
  const int lh   = lane >> 4;
  const int n    = m;
  const size_t row = (size_t)row0 + m;

  // ---- stage 1: s = v * log1p(|v|)/|v| ; inp = crelu(s @ w_in + b_in) -> rbuf (C layout)
  float2 sv[3];
  {
    const float* srcs[3] = {px, pe1, pe2};
#pragma unroll
    for (int j = 0; j < 3; ++j) {
      const float2 v = *(const float2*)(srcs[j] + 2*row);
      const float mag = sqrtf(v.x*v.x + v.y*v.y);
      const float sc  = (mag > 0.f) ? (log1pf(mag)/mag) : 0.f;
      sv[j] = make_float2(v.x*sc, v.y*sc);
    }
  }
#pragma unroll
  for (int e = 0; e < 16; ++e) {
    const int c = lh*16 + e;
    float2 acc = s_bin[c];
#pragma unroll
    for (int j = 0; j < 3; ++j) {
      const float2 wv2 = s_win[j][c];
      acc.x += sv[j].x*wv2.x - sv[j].y*wv2.y;
      acc.y += sv[j].x*wv2.y + sv[j].y*wv2.x;
    }
    acc.x = fmaxf(acc.x, 0.f);
    acc.y = fmaxf(acc.y, 0.f);
    rbuf[wv][m][c] = acc;
  }
  // prefetch layer-1 hidden state while layer 0 runs (global_prefetch_b8)
  __builtin_prefetch(ph + 2*(((size_t)NROWS + row0)*HDIM) + lane*32, 0, 3);

  v16h a_ir, a_ii;
  for (int l = 0; l < NLAYER; ++l) {
    wait_lds();
    buildA(rbuf[wv], lane, &a_ir, &a_ii);      // A_inp (f16)

    // stage h[l] (16 rows x 32 complex) -> hbuf via coalesced b128 loads
    const float* hsrc = ph + 2*(((size_t)l*NROWS + row0)*HDIM);
#pragma unroll
    for (int i = 0; i < 8; ++i) {
      const int f4 = lane + 32*i;
      const float4 hv4 = *(const float4*)(hsrc + 4*f4);
      const int mm = f4 >> 4;
      const int c0 = (f4 & 15) << 1;
      hbuf[wv][mm][c0]   = make_float2(hv4.x, hv4.y);
      hbuf[wv][mm][c0+1] = make_float2(hv4.z, hv4.w);
    }
    wait_lds();
    v16h a_hr, a_hi;
    buildA(hbuf[wv], lane, &a_hr, &a_hi);      // A_h (f16)

    v8f z_re[2], z_im[2];
#pragma unroll
    for (int ct = 0; ct < 4; ++ct) {           // z tiles (0,1), r tiles (2,3)
      const float2 bb = s_gb[l][ct*16 + n];
      v8f cr, ci;
#pragma unroll
      for (int q = 0; q < 8; ++q) { cr[q] = bb.x; ci[q] = bb.y; }
      { const int T = l*6 + ct;                // inp @ gru_wi[:,ct]
        const v16h br = wt[T][0][lane], bi = wt[T][1][lane], bni = wt[T][2][lane];
        cr = wmma_cf(a_ir, br, cr); cr = wmma_cf(a_ii, bni, cr);
        ci = wmma_cf(a_ir, bi, ci); ci = wmma_cf(a_ii, br, ci); }
      { const int T = 12 + l*6 + ct;           // h @ gru_wh[:,ct]
        const v16h br = wt[T][0][lane], bi = wt[T][1][lane], bni = wt[T][2][lane];
        cr = wmma_cf(a_hr, br, cr); cr = wmma_cf(a_hi, bni, cr);
        ci = wmma_cf(a_hr, bi, ci); ci = wmma_cf(a_hi, br, ci); }
#pragma unroll
      for (int q = 0; q < 8; ++q) {            // csigmoid
        cr[q] = 1.f/(1.f + __expf(-cr[q]));
        ci[q] = 1.f/(1.f + __expf(-ci[q]));
      }
      if (ct < 2) { z_re[ct] = cr; z_im[ct] = ci; }
      else {                                   // r (.) h -> rbuf
        const int col = (ct-2)*16 + n;
#pragma unroll
        for (int q = 0; q < 8; ++q) {
          const int mr = q + 8*lh;
          const float2 hv = hbuf[wv][mr][col];
          rbuf[wv][mr][col] = make_float2(cr[q]*hv.x - ci[q]*hv.y,
                                          cr[q]*hv.y + ci[q]*hv.x);
        }
      }
    }
    wait_lds();
    buildA(rbuf[wv], lane, &a_hr, &a_hi);      // A_rh (f16), reuse regs

    float* hn_out = pout + 2*(size_t)NROWS;
#pragma unroll
    for (int ct = 4; ct < 6; ++ct) {           // candidate tiles + h update
      const float2 bb = s_gb[l][ct*16 + n];
      v8f cr, ci;
#pragma unroll
      for (int q = 0; q < 8; ++q) { cr[q] = bb.x; ci[q] = bb.y; }
      { const int T = l*6 + ct;
        const v16h br = wt[T][0][lane], bi = wt[T][1][lane], bni = wt[T][2][lane];
        cr = wmma_cf(a_ir, br, cr); cr = wmma_cf(a_ii, bni, cr);
        ci = wmma_cf(a_ir, bi, ci); ci = wmma_cf(a_ii, br, ci); }
      { const int T = 12 + l*6 + ct;
        const v16h br = wt[T][0][lane], bi = wt[T][1][lane], bni = wt[T][2][lane];
        cr = wmma_cf(a_hr, br, cr); cr = wmma_cf(a_hi, bni, cr);
        ci = wmma_cf(a_hr, bi, ci); ci = wmma_cf(a_hi, br, ci); }
      const int col = (ct-4)*16 + n;
#pragma unroll
      for (int q = 0; q < 8; ++q) {
        const int mr = q + 8*lh;
        const float are = tanhf(cr[q]);        // ctanh
        const float aim = tanhf(ci[q]);
        const float2 hv = hbuf[wv][mr][col];
        const float zr = (ct == 4) ? z_re[0][q] : z_re[1][q];
        const float zi = (ct == 4) ? z_im[0][q] : z_im[1][q];
        const float dre = are - hv.x, dim = aim - hv.y;
        const float2 hn = make_float2(hv.x + zr*dre - zi*dim,
                                      hv.y + zr*dim + zi*dre);
        rbuf[wv][mr][col] = hn;                // next layer input
        *(float2*)(hn_out + 2*(((size_t)l*NROWS + row0 + mr)*HDIM + col)) = hn;
      }
    }
  }

  // ---- output head: crelu(h2 @ w_out1 + b_out1) @ w_out2 + b_out2, * -LAM1
  wait_lds();
  buildA(rbuf[wv], lane, &a_ir, &a_ii);
  v8f pr, pi;
#pragma unroll
  for (int q = 0; q < 8; ++q) { pr[q] = 0.f; pi[q] = 0.f; }
#pragma unroll
  for (int ct = 0; ct < 2; ++ct) {
    const float2 bb = s_bo1[ct*16 + n];
    v8f cr, ci;
#pragma unroll
    for (int q = 0; q < 8; ++q) { cr[q] = bb.x; ci[q] = bb.y; }
    const int T = 24 + ct;
    const v16h br = wt[T][0][lane], bi = wt[T][1][lane], bni = wt[T][2][lane];
    cr = wmma_cf(a_ir, br, cr); cr = wmma_cf(a_ii, bni, cr);
    ci = wmma_cf(a_ir, bi, ci); ci = wmma_cf(a_ii, br, ci);
    const float2 w2v = s_w2[ct*16 + n];
#pragma unroll
    for (int q = 0; q < 8; ++q) {
      const float ore = fmaxf(cr[q], 0.f);     // crelu
      const float oim = fmaxf(ci[q], 0.f);
      pr[q] += ore*w2v.x - oim*w2v.y;          // partial (o @ w_out2)
      pi[q] += ore*w2v.y + oim*w2v.x;
    }
  }
#pragma unroll
  for (int q = 0; q < 8; ++q) rbuf[wv][q + 8*lh][n] = make_float2(pr[q], pi[q]);
  wait_lds();
  if (lane < 16) {
    float2 acc = s_bo2;
#pragma unroll
    for (int q = 0; q < 16; ++q) { const float2 t2 = rbuf[wv][lane][q]; acc.x += t2.x; acc.y += t2.y; }
    *(float2*)(pout + 2*((size_t)row0 + lane)) = make_float2(-LAM1*acc.x, -LAM1*acc.y);
  }
}

extern "C" void kernel_launch(void* const* d_in, const int* in_sizes, int n_in,
                              void* d_out, int out_size, void* d_ws, size_t ws_size,
                              hipStream_t stream) {
  (void)in_sizes; (void)n_in; (void)out_size; (void)d_ws; (void)ws_size;
  const float* px   = (const float*)d_in[0];   // x      (N complex)
  const float* pe1  = (const float*)d_in[1];   // e1
  const float* pe2  = (const float*)d_in[2];   // e2
  const float* ph   = (const float*)d_in[3];   // h      (NL,N,H complex)
  const float* pwin = (const float*)d_in[4];   // w_in   (3,H)
  const float* pbin = (const float*)d_in[5];   // b_in   (H)
  const float* pw1  = (const float*)d_in[6];   // w_out1 (H,H)
  const float* pbo1 = (const float*)d_in[7];   // b_out1 (H)
  const float* pw2  = (const float*)d_in[8];   // w_out2 (H,1)
  const float* pbo2 = (const float*)d_in[9];   // b_out2 (1)
  const float* pgwi = (const float*)d_in[10];  // gru_wi (NL,H,3H)
  const float* pgwh = (const float*)d_in[11];  // gru_wh (NL,H,3H)
  const float* pgb  = (const float*)d_in[12];  // gru_b  (NL,3H)
  float* pout = (float*)d_out;                 // [out (2N floats)][new_h (2*NL*N*H floats)]

  dim3 grid(NROWS / RPB);                      // 2048 blocks x 256 threads (8 waves x 16 rows)
  dim3 block(WPB * 32);
  gru_pipeline_kernel<<<grid, block, 0, stream>>>(px, pe1, pe2, ph, pwin, pbin,
                                                  pw1, pbo1, pw2, pbo2, pgwi, pgwh,
                                                  pgb, pout);
}
